// SimpleNetwork_80135499809327
// MI455X (gfx1250) — matrix operand order repaired and checked by reading, not compiled, not run on hardware
//
#include <hip/hip_runtime.h>
#include <hip/hip_bf16.h>
#include <stdint.h>

typedef __attribute__((ext_vector_type(16))) __bf16        v16bf;
typedef __attribute__((ext_vector_type(8)))  float         v8f;
typedef __attribute__((ext_vector_type(8)))  unsigned int  v8u;

#define D_FEAT 256
#define D_U32  128          // D_FEAT/2 packed-bf16 words per row
#define N_ROWS 8192
#define KNB    32
#define TAU_F  0.07f
#define FS     257          // padded LDS row stride (floats), kernel 1
#define BSTR   133          // padded LDS row stride (u32 words), kernel 2 fyn tiles

__device__ __forceinline__ unsigned int pk_bf16(float a, float b) {
  unsigned int ua = __float_as_uint(a);
  unsigned int ub = __float_as_uint(b);
  ua += 0x7FFFu + ((ua >> 16) & 1u);   // round-to-nearest-even bf16
  ub += 0x7FFFu + ((ub >> 16) & 1u);
  return (ua >> 16) | (ub & 0xFFFF0000u);
}

__device__ __forceinline__ unsigned long long shfl_xor_u64(unsigned long long v, int m) {
  unsigned int lo = (unsigned int)v, hi = (unsigned int)(v >> 32);
  lo = __shfl_xor(lo, m, 32);
  hi = __shfl_xor(hi, m, 32);
  return ((unsigned long long)hi << 32) | lo;
}

// Async DMA: one tile row (32 lanes x 16B = 512B) per instruction; 16 rows/tile.
// Global rows are dense (512B); LDS rows are padded to BSTR*4 bytes.
__device__ __forceinline__ void issue_tile_async(unsigned lds_lane_base,
                                                 const unsigned int* g_lane_base) {
  #pragma unroll
  for (int r = 0; r < 16; ++r) {
    asm volatile("global_load_async_to_lds_b128 %0, %1, off"
                 :: "v"(lds_lane_base + r * (BSTR * 4)),
                    "v"(g_lane_base + r * D_U32)
                 : "memory");
  }
}
__device__ __forceinline__ void wait_async_prev16() {
  asm volatile("s_wait_asynccnt 0x10" ::: "memory");   // <=16 outstanding
}
__device__ __forceinline__ void wait_async_all() {
  asm volatile("s_wait_asynccnt 0x0" ::: "memory");
}

// ---------------------------------------------------------------------------
// Kernel 1: f = rows @ W^T + b, L2-normalize rows, emit packed bf16 features.
// ---------------------------------------------------------------------------
__global__ __launch_bounds__(128)
void feat_kernel(const float* __restrict__ x, const float* __restrict__ y,
                 const float* __restrict__ W, const float* __restrict__ b,
                 unsigned int* __restrict__ fn_u) {
  __shared__ float ftile[4][16 * FS];   // ~64 KB

  const int tid = threadIdx.x;
  const int wv  = tid >> 5;
  const int l   = tid & 31;
  const int m   = l & 15;
  const int h   = l >> 4;

  const int row0 = blockIdx.x * 64 + wv * 16;
  const int grow = row0 + m;
  const float* src = (grow < N_ROWS) ? (x + (size_t)grow * D_FEAT)
                                     : (y + (size_t)(grow - N_ROWS) * D_FEAT);

  // A fragments: koff(j,h) = (j<4 ? 2j : 8+2j) + 8h
  v16bf afrag[8];
  #pragma unroll
  for (int ks = 0; ks < 8; ++ks) {
    v8u au;
    #pragma unroll
    for (int j = 0; j < 8; ++j) {
      int pidx = ks * 16 + ((j < 4) ? j : 4 + j) + 4 * h;
      float2 v = ((const float2*)src)[pidx];
      au[j] = pk_bf16(v.x, v.y);
    }
    afrag[ks] = __builtin_bit_cast(v16bf, au);
  }

  float* ft = ftile[wv];

  for (int nt = 0; nt < 16; ++nt) {
    const int nbase = nt * 16;
    const float* wr = W + (size_t)(nbase + m) * D_FEAT;   // B[k][n] = W[n][k]
    v8f acc = {};
    #pragma unroll
    for (int ks = 0; ks < 8; ++ks) {
      v8u bu;
      #pragma unroll
      for (int j = 0; j < 8; ++j) {
        float2 v = ((const float2*)wr)[ks * 16 + ((j < 4) ? j : 4 + j) + 4 * h];
        bu[j] = pk_bf16(v.x, v.y);
      }
      v16bf bfrag = __builtin_bit_cast(v16bf, bu);
      acc = __builtin_amdgcn_wmma_f32_16x16x32_bf16(false, afrag[ks], false, bfrag,
                                                    (short)0, acc, false, false);
    }
    const float bias = b[nbase + m];
    #pragma unroll
    for (int v = 0; v < 8; ++v)
      ft[(v + 8 * h) * FS + nbase + m] = acc[v] + bias;
  }
  __syncthreads();

  // Row norms: lanes (m, m+16) split row m; combine with shfl_xor(16).
  float ss = 0.f;
  const float* rowp = ft + m * FS + h * 128;
  #pragma unroll 8
  for (int c = 0; c < 128; ++c) { float v = rowp[c]; ss += v * v; }
  ss += __shfl_xor(ss, 16, 32);
  const float scale = rsqrtf(ss + 1e-8f);

  unsigned int* dst = fn_u + (size_t)grow * D_U32 + h * 64;
  #pragma unroll 4
  for (int i = 0; i < 64; ++i)
    dst[i] = pk_bf16(rowp[2 * i] * scale, rowp[2 * i + 1] * scale);
}

// ---------------------------------------------------------------------------
// Kernel 2: fused sim = fxn @ fyn^T with streaming per-row top-32 + softmax.
// fyn tiles double-buffered in LDS via GLOBAL_LOAD_ASYNC_TO_LDS_B128.
// ---------------------------------------------------------------------------
__global__ __launch_bounds__(128)
void simtopk_kernel(const unsigned int* __restrict__ fxn_u,
                    const unsigned int* __restrict__ fyn_u,
                    float* __restrict__ out_soft,
                    int*   __restrict__ out_idx) {
  __shared__ unsigned int fx_s[16 * D_U32];     // 8 KB   fxn tile
  __shared__ unsigned int yb[4][2][16 * BSTR];  // ~68 KB fyn double buffers
  __shared__ float sim_s[4][16 * 17];           // per-wave sim tiles
  __shared__ float mrgv[128 * KNB];             // 16 KB
  __shared__ int   mrgi[128 * KNB];             // 16 KB

  const int tid = threadIdx.x;
  const int wv  = tid >> 5;
  const int l   = tid & 31;
  const int m   = l & 15;
  const int h   = l >> 4;
  const int rbase = blockIdx.x * 16;

  for (int i = tid; i < 16 * D_U32; i += 128)
    fx_s[i] = fxn_u[(size_t)rbase * D_U32 + i];

  // Kick off this wave's first fyn tile while we build A fragments.
  const unsigned lb0 = (unsigned)(size_t)&yb[wv][0][l * 4];
  const unsigned lb1 = (unsigned)(size_t)&yb[wv][1][l * 4];
  issue_tile_async(lb0, fyn_u + (size_t)wv * 16 * D_U32 + l * 4);

  __syncthreads();

  v16bf afrag[8];
  #pragma unroll
  for (int ks = 0; ks < 8; ++ks) {
    v8u au;
    #pragma unroll
    for (int j = 0; j < 8; ++j)
      au[j] = fx_s[m * D_U32 + ks * 16 + ((j < 4) ? j : 4 + j) + 4 * h];
    afrag[ks] = __builtin_bit_cast(v16bf, au);
  }

  float tv[KNB]; int ti[KNB];
  #pragma unroll
  for (int q = 0; q < KNB; ++q) { tv[q] = -3.0e38f; ti[q] = 0; }
  float minval = -3.0e38f; int minpos = 0;

  float* st = sim_s[wv];
  const unsigned int* lptr[2] = { &yb[wv][0][0], &yb[wv][1][0] };
  int pb = 0;

  for (int ct = wv; ct < 512; ct += 4) {
    const int jbase = ct * 16;
    const int nct = ct + 4;
    if (nct < 512) {                     // prefetch next tile into other buffer
      issue_tile_async(pb ? lb0 : lb1, fyn_u + (size_t)nct * 16 * D_U32 + l * 4);
      wait_async_prev16();               // current tile's 16 ops done (in-order)
    } else {
      wait_async_all();
    }

    const unsigned int* bb = lptr[pb];
    v8f acc = {};
    #pragma unroll
    for (int ks = 0; ks < 8; ++ks) {
      v8u bu;
      #pragma unroll
      for (int j = 0; j < 8; ++j)
        bu[j] = bb[m * BSTR + ks * 16 + ((j < 4) ? j : 4 + j) + 4 * h];   // conflict-free (stride 133)
      v16bf bfrag = __builtin_bit_cast(v16bf, bu);
      acc = __builtin_amdgcn_wmma_f32_16x16x32_bf16(false, afrag[ks], false, bfrag,
                                                    (short)0, acc, false, false);
    }
    #pragma unroll
    for (int v = 0; v < 8; ++v)
      st[(v + 8 * h) * 17 + m] = acc[v];

    // Lane scans row m, columns h*8 .. h*8+7.
    #pragma unroll
    for (int s = 0; s < 8; ++s) {
      const int n = h * 8 + s;
      const float val = st[m * 17 + n];
      if (val > minval) {
        const int gidx = jbase + n;
        #pragma unroll
        for (int q = 0; q < KNB; ++q)
          if (q == minpos) { tv[q] = val; ti[q] = gidx; }
        minval = tv[0]; minpos = 0;
        #pragma unroll
        for (int q = 1; q < KNB; ++q)
          if (tv[q] < minval) { minval = tv[q]; minpos = q; }
      }
    }
    pb ^= 1;
  }

  {
    const int slot = wv * 32 + l;
    #pragma unroll
    for (int q = 0; q < KNB; ++q) {
      mrgv[slot * KNB + q] = tv[q];
      mrgi[slot * KNB + q] = ti[q];
    }
  }
  __syncthreads();

  // Final merge: wave wv handles rows wv, wv+4, wv+8, wv+12.
  for (int rr = 0; rr < 4; ++rr) {
    const int r = rr * 4 + wv;
    unsigned long long key[8];
    #pragma unroll
    for (int s = 0; s < 8; ++s) {
      const int slot = (s & 3) * 32 + r + (s >> 2) * 16;
      const float v = mrgv[slot * KNB + l];
      const int   g = mrgi[slot * KNB + l];
      unsigned int u = __float_as_uint(v);
      u = (u & 0x80000000u) ? ~u : (u | 0x80000000u);   // order-preserving map
      key[s] = ((unsigned long long)u << 32) | (unsigned int)(~g); // low idx wins
    }
    float selv = 0.f; int seli = 0;
    for (int t = 0; t < KNB; ++t) {
      unsigned long long best = key[0];
      #pragma unroll
      for (int s = 1; s < 8; ++s) best = (key[s] > best) ? key[s] : best;
      #pragma unroll
      for (int d = 16; d >= 1; d >>= 1) {
        unsigned long long o = shfl_xor_u64(best, d);
        best = (o > best) ? o : best;
      }
      const unsigned int u = (unsigned int)(best >> 32);
      const float wvv = (u & 0x80000000u) ? __uint_as_float(u & 0x7FFFFFFFu)
                                          : __uint_as_float(~u);
      const int wg = (int)(~(unsigned int)best);
      if (l == t) { selv = wvv; seli = wg; }
      #pragma unroll
      for (int s = 0; s < 8; ++s) if (key[s] == best) key[s] = 0ull;
    }
    const float mx = __shfl(selv, 0, 32);
    float e = __expf((selv - mx) * (1.0f / TAU_F));
    float sum = e;
    #pragma unroll
    for (int d = 16; d >= 1; d >>= 1) sum += __shfl_xor(sum, d, 32);
    const int orow = rbase + r;
    out_soft[(size_t)orow * KNB + l] = e / sum;
    out_idx [(size_t)orow * KNB + l] = seli;
  }
}

extern "C" void kernel_launch(void* const* d_in, const int* in_sizes, int n_in,
                              void* d_out, int out_size, void* d_ws, size_t ws_size,
                              hipStream_t stream) {
  const float* x = (const float*)d_in[0];
  const float* y = (const float*)d_in[1];
  const float* W = (const float*)d_in[2];
  const float* b = (const float*)d_in[3];

  unsigned int* fn_u = (unsigned int*)d_ws;                 // 16384*128 u32 = 8 MB
  const unsigned int* fxn_u = fn_u;
  const unsigned int* fyn_u = fn_u + (size_t)N_ROWS * D_U32;

  float* out_soft = (float*)d_out;                          // [8192,32] f32
  int*   out_idx  = (int*)((float*)d_out + (size_t)N_ROWS * KNB); // [8192,32] i32

  feat_kernel<<<dim3(256), dim3(128), 0, stream>>>(x, y, W, b, fn_u);
  simtopk_kernel<<<dim3(512), dim3(128), 0, stream>>>(fxn_u, fyn_u, out_soft, out_idx);
}
